// MultiHeadAttention_84250078478822
// MI455X (gfx1250) — compile-verified, matrix operand order checked by reading, not run on hardware
//
#include <hip/hip_runtime.h>

// ---------------------------------------------------------------------------
// MultiHeadAttention for MI455X (gfx1250, wave32, WMMA bf16 16x16x32)
// B=4, S=2048, D=1024, H=16, DK=DV=64
// ---------------------------------------------------------------------------

typedef __attribute__((ext_vector_type(16))) __bf16         v16bf;
typedef __attribute__((ext_vector_type(8)))  float          v8f;
typedef __attribute__((ext_vector_type(8)))  unsigned short v8us;
typedef int v4i __attribute__((vector_size(16)));   // matches async-LDS builtin param

union BF16x16 { v16bf v; v8us h[2]; };

// native single-op conversion (gfx1250 has bf16 VALU support)
__device__ __forceinline__ unsigned short f2bf(float f) {
    return __builtin_bit_cast(unsigned short, (__bf16)f);
}

#define S_LEN 2048
#define D_DIM 1024
#define H_NUM 16
#define N_DIM 64

// ---- CDNA5 async global->LDS path (builtin confirmed present) ----
#if defined(__has_builtin)
#if __has_builtin(__builtin_amdgcn_global_load_async_to_lds_b128)
#define HAVE_ASYNC_LDS 1
#endif
#if __has_builtin(__builtin_amdgcn_s_wait_asynccnt)
#define HAVE_WAIT_ASYNC_BUILTIN 1
#endif
#endif

__device__ __forceinline__ void wait_asynccnt0() {
#if defined(HAVE_WAIT_ASYNC_BUILTIN)
    __builtin_amdgcn_s_wait_asynccnt(0);
#else
    asm volatile("s_wait_asynccnt 0x0" ::: "memory");
#endif
}

// ---------------------------------------------------------------------------
// Flash attention: per (b,h), online softmax over key tiles of 32.
// block = 128 threads (4 waves); each wave owns 16 query rows.
// K tile staged to LDS via async global->LDS (ASYNCcnt-tracked) and shared by
// all waves; V tile staged transposed for the PV B-fragments.
// (Placed first in the TU so the disasm snippet shows this kernel.)
// ---------------------------------------------------------------------------
#define KTILE 32
#define AWAVE 4
#define LDK   72   // 144B rows: 16B aligned, breaks 128B bank-conflict pattern
#define LDV   40

__global__ __launch_bounds__(128) void flash_attn(
    const unsigned short* __restrict__ Q,   // [B,H,S,64] bf16
    const unsigned short* __restrict__ K,   // [B,H,S,64] bf16
    const unsigned short* __restrict__ V,   // [B,H,S,64] bf16
    float* __restrict__ Out)                // [B,S,H*64] f32
{
    const int bh   = blockIdx.x;
    const int b    = bh / H_NUM;
    const int h    = bh % H_NUM;
    const int q0   = blockIdx.y * (AWAVE * 16);
    const int tid  = threadIdx.x;
    const int lane = tid & 31;
    const int wave = tid >> 5;
    const int hi   = lane >> 4;
    const int lm   = lane & 15;

    __shared__ alignas(16) unsigned short Kt[KTILE * LDK];          // K tile row-major
    __shared__ alignas(16) unsigned short Vt[N_DIM * LDV];          // V tile transposed
    __shared__ float          Sf[AWAVE][16][KTILE];                 // raw scores
    __shared__ alignas(16) unsigned short Pl[AWAVE][16][KTILE];     // probs bf16
    __shared__ float m_s[AWAVE][16], l_s[AWAVE][16], c_s[AWAVE][16];

    const unsigned short* Qbh = Q + (((long)b * H_NUM + h) * S_LEN) * N_DIM;
    const unsigned short* Kbh = K + (((long)b * H_NUM + h) * S_LEN) * N_DIM;
    const unsigned short* Vbh = V + (((long)b * H_NUM + h) * S_LEN) * N_DIM;

    // Q fragments (A layout) for this wave's 16 rows, K-dim 0..31 and 32..63
    const long qr = q0 + wave * 16 + lm;
    BF16x16 qf[2];
    qf[0].h[0] = *(const v8us*)&Qbh[qr * N_DIM +      hi * 8];
    qf[0].h[1] = *(const v8us*)&Qbh[qr * N_DIM + 16 + hi * 8];
    qf[1].h[0] = *(const v8us*)&Qbh[qr * N_DIM + 32 + hi * 8];
    qf[1].h[1] = *(const v8us*)&Qbh[qr * N_DIM + 48 + hi * 8];

    v8f o[4];
#pragma unroll
    for (int i = 0; i < 4; ++i) o[i] = (v8f){};
    if (lane < 16) { m_s[wave][lane] = -1e30f; l_s[wave][lane] = 0.f; }

    for (int t = 0; t < S_LEN; t += KTILE) {
        __syncthreads();
        // ---- stage K tile [32][64] row-major into LDS (16 elems/thread) ----
        {
            const int e0 = tid * 16;
            const int kk = e0 >> 6;               // row in tile
            const int c0 = e0 & 63;               // col 0/16/32/48
            const unsigned short* src = Kbh + (long)(t + kk) * N_DIM + c0;
            unsigned short* dst = &Kt[kk * LDK + c0];
#if defined(HAVE_ASYNC_LDS)
            __builtin_amdgcn_global_load_async_to_lds_b128(
                (v4i*)src, (v4i*)dst, 0, 0);
            __builtin_amdgcn_global_load_async_to_lds_b128(
                (v4i*)(src + 8), (v4i*)(dst + 8), 0, 0);
#else
            *(v8us*)dst       = *(const v8us*)src;
            *(v8us*)(dst + 8) = *(const v8us*)(src + 8);
#endif
        }
        // ---- stage V tile transposed: V[t+kk][n] -> Vt[n][kk] ----
        {
            const int e0 = tid * 16;
            const int kk = e0 >> 6;
            const int n0 = e0 & 63;
            const unsigned short* src = Vbh + (long)(t + kk) * N_DIM + n0;
#pragma unroll
            for (int j = 0; j < 16; ++j) Vt[(n0 + j) * LDV + kk] = src[j];
        }
        // prefetch next iteration's K/V tiles while this tile is consumed
        if (t + KTILE < S_LEN) {
            const int e0 = tid * 16;
            const int kk = e0 >> 6;
            const int c0 = e0 & 63;
            __builtin_prefetch(Kbh + (long)(t + KTILE + kk) * N_DIM + c0, 0, 0);
            __builtin_prefetch(Vbh + (long)(t + KTILE + kk) * N_DIM + c0, 0, 0);
        }
#if defined(HAVE_ASYNC_LDS)
        wait_asynccnt0();
#endif
        __syncthreads();

        // scores S = Q . K^T for 32 keys (two 16-key subtiles, 2 WMMAs each)
#pragma unroll
        for (int sub = 0; sub < 2; ++sub) {
            const int krow = sub * 16 + lm;       // B-matrix column = key row
            BF16x16 kf0, kf1;
            kf0.h[0] = *(const v8us*)&Kt[krow * LDK +      hi * 8];
            kf0.h[1] = *(const v8us*)&Kt[krow * LDK + 16 + hi * 8];
            kf1.h[0] = *(const v8us*)&Kt[krow * LDK + 32 + hi * 8];
            kf1.h[1] = *(const v8us*)&Kt[krow * LDK + 48 + hi * 8];
            v8f s = (v8f){};
            s = __builtin_amdgcn_wmma_f32_16x16x32_bf16(
                false, qf[0].v, false, kf0.v, (short)0, s, false, false);
            s = __builtin_amdgcn_wmma_f32_16x16x32_bf16(
                false, qf[1].v, false, kf1.v, (short)0, s, false, false);
#pragma unroll
            for (int i = 0; i < 8; ++i)
                Sf[wave][i + 8 * hi][sub * 16 + lm] = s[i];
        }

        // online softmax: lanes 0..15 each own one query row
        if (lane < 16) {
            const int r = lane;
            float mo = m_s[wave][r];
            float mn = mo;
#pragma unroll
            for (int j = 0; j < KTILE; ++j) mn = fmaxf(mn, Sf[wave][r][j]);
            const float c = __expf(mo - mn);
            float sum = 0.f;
#pragma unroll
            for (int j = 0; j < KTILE; ++j) {
                const float p = __expf(Sf[wave][r][j] - mn);
                sum += p;
                Pl[wave][r][j] = f2bf(p);
            }
            m_s[wave][r] = mn;
            l_s[wave][r] = l_s[wave][r] * c + sum;
            c_s[wave][r] = c;
        }

        // rescale running O by per-row correction
#pragma unroll
        for (int i = 0; i < 8; ++i) {
            const float c = c_s[wave][i + 8 * hi];
#pragma unroll
            for (int nt = 0; nt < 4; ++nt) o[nt][i] *= c;
        }

        // O += P . V   (P is A-matrix 16x32, V fragments from transposed tile)
        BF16x16 pf;
        pf.h[0] = *(const v8us*)&Pl[wave][lm][hi * 8];
        pf.h[1] = *(const v8us*)&Pl[wave][lm][16 + hi * 8];
#pragma unroll
        for (int nt = 0; nt < 4; ++nt) {
            const int n = nt * 16 + lm;
            BF16x16 vf;
            vf.h[0] = *(const v8us*)&Vt[n * LDV + hi * 8];
            vf.h[1] = *(const v8us*)&Vt[n * LDV + 16 + hi * 8];
            o[nt] = __builtin_amdgcn_wmma_f32_16x16x32_bf16(
                false, pf.v, false, vf.v, (short)0, o[nt], false, false);
        }
    }

    // normalize and store into [B, S, H*DV] (head-concatenated layout)
    float inv_l[8];
#pragma unroll
    for (int i = 0; i < 8; ++i) inv_l[i] = 1.f / l_s[wave][i + 8 * hi];
    float* Ob = Out + ((long)b * S_LEN + q0 + wave * 16) * (H_NUM * N_DIM) + h * N_DIM;
#pragma unroll
    for (int nt = 0; nt < 4; ++nt)
#pragma unroll
        for (int i = 0; i < 8; ++i)
            Ob[(long)(i + 8 * hi) * (H_NUM * N_DIM) + nt * 16 + lm] =
                o[nt][i] * inv_l[i];
}

// ---------------------------------------------------------------------------
// Projection GEMM:  Out[b,h,s,n] (bf16) = scale * X[b,s,:] . W[h,:,n]
// block = 256 threads (8 waves), tile 128 rows x 64 cols, K-step 32
// ---------------------------------------------------------------------------
#define PBM 128
#define PKT 32
#define LDA 40   // LDS stride (elements): 80B rows keep 16B-aligned fragments
#define LDB 40

__global__ __launch_bounds__(256) void proj_gemm(
    const float* __restrict__ X,          // [B,S,D] f32
    const float* __restrict__ W,          // [H,D,64] f32
    unsigned short* __restrict__ Out,     // [B,H,S,64] bf16 bits
    float scale)
{
    const int bh   = blockIdx.x;
    const int b    = bh / H_NUM;
    const int h    = bh % H_NUM;
    const int row0 = blockIdx.y * PBM;
    const int tid  = threadIdx.x;
    const int lane = tid & 31;
    const int wave = tid >> 5;            // 0..7
    const int hi   = lane >> 4;
    const int lm   = lane & 15;

    __shared__ alignas(16) unsigned short As[PBM * LDA];
    __shared__ alignas(16) unsigned short Bt[N_DIM * LDB];

    const float* Xb = X + ((long)b * S_LEN + row0) * D_DIM;
    const float* Wh = W + (long)h * D_DIM * N_DIM;

    v8f acc[4];
#pragma unroll
    for (int i = 0; i < 4; ++i) acc[i] = (v8f){};

    for (int k0 = 0; k0 < D_DIM; k0 += PKT) {
        __syncthreads();
        // stage A: 128x32 f32 -> bf16 LDS, 16 elems/thread, packed 16B stores
        {
            const int r  = tid >> 1;
            const int c0 = (tid & 1) * 16;
            const float* src = Xb + (long)r * D_DIM + k0 + c0;
            v8us p0, p1;
#pragma unroll
            for (int j = 0; j < 8; ++j) { p0[j] = f2bf(src[j]); p1[j] = f2bf(src[j + 8]); }
            *(v8us*)&As[r * LDA + c0]     = p0;
            *(v8us*)&As[r * LDA + c0 + 8] = p1;
        }
        // stage B transposed: W[h, k0+kk, n] -> Bt[n][kk]  (8 elems/thread)
        {
            const int e0 = tid * 8;
            const int kk = e0 >> 6;       // 0..31
            const int n0 = e0 & 63;
            const float* src = Wh + (long)(k0 + kk) * N_DIM + n0;
#pragma unroll
            for (int j = 0; j < 8; ++j) Bt[(n0 + j) * LDB + kk] = f2bf(src[j]);
        }
        __syncthreads();

        // A fragment (ISA 16-bit A 16x32 layout): two ds_load_b128 per lane
        const int ar = wave * 16 + lm;
        BF16x16 a;
        a.h[0] = *(const v8us*)&As[ar * LDA + hi * 8];
        a.h[1] = *(const v8us*)&As[ar * LDA + 16 + hi * 8];

#pragma unroll
        for (int nt = 0; nt < 4; ++nt) {
            const int bn = nt * 16 + lm;
            BF16x16 bb;
            bb.h[0] = *(const v8us*)&Bt[bn * LDB + hi * 8];
            bb.h[1] = *(const v8us*)&Bt[bn * LDB + 16 + hi * 8];
            acc[nt] = __builtin_amdgcn_wmma_f32_16x16x32_bf16(
                false, a.v, false, bb.v, (short)0, acc[nt], false, false);
        }
    }

    // store (C layout: lane owns col lm, rows i + 8*hi)
    unsigned short* Ob =
        Out + (((long)b * H_NUM + h) * S_LEN + row0 + wave * 16) * N_DIM;
#pragma unroll
    for (int nt = 0; nt < 4; ++nt)
#pragma unroll
        for (int i = 0; i < 8; ++i)
            Ob[(long)(i + 8 * hi) * N_DIM + nt * 16 + lm] = f2bf(acc[nt][i] * scale);
}

// ---------------------------------------------------------------------------
// launcher
// ---------------------------------------------------------------------------
extern "C" void kernel_launch(void* const* d_in, const int* in_sizes, int n_in,
                              void* d_out, int out_size, void* d_ws, size_t ws_size,
                              hipStream_t stream) {
    (void)in_sizes; (void)n_in; (void)out_size; (void)ws_size;
    const int B = 4;

    const float* X  = (const float*)d_in[0];
    const float* Wq = (const float*)d_in[1];
    const float* Wk = (const float*)d_in[2];
    const float* Wv = (const float*)d_in[3];
    float* out = (float*)d_out;

    const size_t perBuf = (size_t)B * H_NUM * S_LEN * N_DIM;   // 8M bf16 elems
    unsigned short* Qb = (unsigned short*)d_ws;
    unsigned short* Kb = Qb + perBuf;
    unsigned short* Vb = Kb + perBuf;

    dim3 gridP(B * H_NUM, S_LEN / PBM);
    proj_gemm<<<gridP, 256, 0, stream>>>(X, Wq, Qb, 0.125f);  // fold 1/sqrt(DK) into Q
    proj_gemm<<<gridP, 256, 0, stream>>>(X, Wk, Kb, 1.0f);
    proj_gemm<<<gridP, 256, 0, stream>>>(X, Wv, Vb, 1.0f);

    dim3 gridA(B * H_NUM, S_LEN / (AWAVE * 16));
    flash_attn<<<gridA, 128, 0, stream>>>(Qb, Kb, Vb, out);
}
